// MPNN_2903397893033
// MI455X (gfx1250) — matrix operand order repaired and checked by reading, not compiled, not run on hardware
//
#include <hip/hip_runtime.h>
#include <hip/hip_bf16.h>
#include <stdint.h>

// MPNN: B=8, N=4096, F=U=128
#define Bn 8
#define Nn 4096
#define Fn 128
#define Un 128

typedef __attribute__((ext_vector_type(16))) __bf16 v16bf;
typedef __attribute__((ext_vector_type(8)))  float  v8f;
typedef __attribute__((ext_vector_type(4)))  int    v4i;

union V16 { uint4 q[2]; v16bf v; };

// LDS row strides in bf16 elements.
// 72 elems = 144 B = 36 dwords: 36*k mod 64 distinct for k=0..15 -> conflict-free frag reads.
#define MT2_S 72    // mask tile rows [32 j][64 k] padded to 72
#define XT2_S 72    // x    tile rows [128 f][64 k] padded to 72
#define ME_S  136   // mean tile rows [32 j][128 f] padded to 136

// ---------------------------------------------------------------------------
// prep_x: x fp32 [B][N][F] -> xh bf16 [B][N][F] and xT bf16 [B][F][N]
// grid = B*(N/32)*(F/32) = 4096 blocks, 256 threads
// ---------------------------------------------------------------------------
__global__ __launch_bounds__(256) void prep_x_kernel(const float* __restrict__ x,
                                                     __bf16* __restrict__ xh,
                                                     __bf16* __restrict__ xT)
{
    __shared__ __bf16 tl[32][33];   // [n_local][f_local]
    const int blk = blockIdx.x;
    const int bb  = blk >> 9;
    const int rem = blk & 511;
    const int n0  = (rem >> 2) << 5;
    const int f0  = (rem & 3) << 5;
    const int t = threadIdx.x;
    const int r = t >> 3, c = (t & 7) << 2;

    const size_t gidx = ((size_t)(bb * Nn + n0 + r) * Fn) + f0 + c;
    const float4 v = *(const float4*)(x + gidx);
    const __bf16 b0 = (__bf16)v.x, b1 = (__bf16)v.y, b2 = (__bf16)v.z, b3 = (__bf16)v.w;
    xh[gidx + 0] = b0; xh[gidx + 1] = b1; xh[gidx + 2] = b2; xh[gidx + 3] = b3;
    tl[r][c + 0] = b0; tl[r][c + 1] = b1; tl[r][c + 2] = b2; tl[r][c + 3] = b3;
    __syncthreads();

    const int fr = t >> 3, cc = (t & 7) << 2;
    const size_t tbase = ((size_t)(bb * Fn + f0 + fr) * Nn) + n0 + cc;
    xT[tbase + 0] = tl[cc + 0][fr];
    xT[tbase + 1] = tl[cc + 1][fr];
    xT[tbase + 2] = tl[cc + 2][fr];
    xT[tbase + 3] = tl[cc + 3][fr];
}

// ---------------------------------------------------------------------------
// prep_w: W fp32 [F][U] -> wT bf16 [U][F]  (both matrices)
// ---------------------------------------------------------------------------
__global__ __launch_bounds__(256) void prep_w_kernel(const float* __restrict__ Wm,
                                                     const float* __restrict__ Wu,
                                                     __bf16* __restrict__ wTm,
                                                     __bf16* __restrict__ wTu)
{
    const int idx  = blockIdx.x * 256 + threadIdx.x;
    const int m    = idx >> 14;
    const int rest = idx & 16383;
    const int u = rest >> 7, f = rest & 127;
    const float* src = m ? Wu  : Wm;
    __bf16*      dst = m ? wTu : wTm;
    dst[u * Fn + f] = (__bf16)src[f * Un + u];
}

// ---------------------------------------------------------------------------
// main: per block (batch, 32-receiver j-tile); 8 waves.
// Stage 1: msum = mask^T @ x via wmma bf16, K=4096 in 64-wide double-buffered
//          steps (adj streamed NT; x tile staged with async global->LDS copy).
// Stage 2: out = relu(mean@Wmsg + x@Wupd + bias), chained into f32 accums.
// grid = B*(N/32) = 1024 blocks, 256 threads
// ---------------------------------------------------------------------------
__global__ __launch_bounds__(256) void mpnn_main(const int*    __restrict__ adj,
                                                 const __bf16* __restrict__ xh,
                                                 const __bf16* __restrict__ xT,
                                                 const __bf16* __restrict__ wTm,
                                                 const __bf16* __restrict__ wTu,
                                                 const float*  __restrict__ bias,
                                                 float*        __restrict__ out)
{
    __shared__ __bf16 maskT[2 * 32 * MT2_S];   // ping-pong [j][k] bf16 {0,1}
    __shared__ __bf16 xt[2 * 128 * XT2_S];     // ping-pong [f][k]
    __shared__ __bf16 meanT[32 * ME_S];        // [j][f]
    __shared__ float  degs[32];

    const int t     = threadIdx.x;
    const int lane  = t & 31;
    const int wave  = t >> 5;
    const int batch = blockIdx.x >> 7;
    const int j0    = (blockIdx.x & 127) << 5;

    const int Msub = wave & 1;     // 16-row j sub-tile
    const int fgrp = wave >> 1;    // pair of 16-wide column tiles

    if (t < 32) degs[t] = 0.0f;

    // staging mapping: adj tile 64k x 32j -> 2 rows x 4 cols per thread
    const int kr = (t >> 3) << 1;     // k rows kr, kr+1
    const int sc = (t & 7) << 2;      // j cols sc..sc+3
    // staging mapping: x tile 128f x 64k -> 64B per thread (async copy)
    const int xr   = t >> 1;          // f row 0..127
    const int xk32 = (t & 1) << 5;    // k offset 0 / 32

    // WMMA fragment addressing (wave32; contiguous k runs, same order for A and B):
    const int klo  = (lane < 16) ? 0 : 16;   // A: run  {klo..klo+15}
    const int bk   = (lane < 16) ? 0 : 16;   // B: run  {bk..bk+15}
    const int ln15 = lane & 15;
    const int mrow = Msub * 16 + ln15;
    const int f0a  = (fgrp * 2 + 0) * 16;
    const int f0b  = f0a + 16;

    float degp[4] = {0.f, 0.f, 0.f, 0.f};
    v8f acc0 = {}; v8f acc1 = {};

    const size_t adj_b = (size_t)batch * Nn * Nn;
    const size_t xT_b  = (size_t)batch * Fn * Nn;

    auto stage_tile = [&](int k0s, int buf) {
        // ---- adj: 2 NT b128 loads, pack k-pairs -> b32 LDS stores ----
        const size_t abase = adj_b + (size_t)(k0s + kr) * Nn + j0 + sc;
        const v4i a0 = __builtin_nontemporal_load((const v4i*)(adj + abase));
        const v4i a1 = __builtin_nontemporal_load((const v4i*)(adj + abase + Nn));
        __bf16* mbuf = maskT + buf * (32 * MT2_S);
        #pragma unroll
        for (int q = 0; q < 4; ++q) {
            const uint32_t d = (a0[q] ? 0x3f80u : 0u) | (a1[q] ? 0x3f800000u : 0u);
            *(uint32_t*)(mbuf + (sc + q) * MT2_S + kr) = d;
            degp[q] += (a0[q] ? 1.f : 0.f) + (a1[q] ? 1.f : 0.f);
        }
        // ---- x tile: pure copy -> async global->LDS (ASYNCcnt) ----
        const __bf16* gsrc = xT + xT_b + (size_t)xr * Nn + k0s + xk32;
        __bf16* ldst = xt + buf * (128 * XT2_S) + (size_t)xr * XT2_S + xk32;
        const uint32_t ldsa = (uint32_t)(uintptr_t)ldst;   // low 32b = LDS offset
        const uint64_t ga   = (uint64_t)(uintptr_t)gsrc;
        asm volatile(
            "global_load_async_to_lds_b128 %0, %1, off offset:0\n"
            "global_load_async_to_lds_b128 %0, %1, off offset:16\n"
            "global_load_async_to_lds_b128 %0, %1, off offset:32\n"
            "global_load_async_to_lds_b128 %0, %1, off offset:48"
            :: "v"(ldsa), "v"(ga) : "memory");
    };

    auto compute_tile = [&](int buf) {
        const __bf16* mbuf = maskT + buf * (32 * MT2_S);
        const __bf16* xbuf = xt + buf * (128 * XT2_S);
        #pragma unroll
        for (int kc = 0; kc < 64; kc += 32) {
            V16 a, b0, b1;
            a.q[0]  = *(const uint4*)(mbuf + mrow * MT2_S + kc + klo);
            a.q[1]  = *(const uint4*)(mbuf + mrow * MT2_S + kc + klo + 8);
            b0.q[0] = *(const uint4*)(xbuf + (f0a + ln15) * XT2_S + kc + bk);
            b0.q[1] = *(const uint4*)(xbuf + (f0a + ln15) * XT2_S + kc + bk + 8);
            b1.q[0] = *(const uint4*)(xbuf + (f0b + ln15) * XT2_S + kc + bk);
            b1.q[1] = *(const uint4*)(xbuf + (f0b + ln15) * XT2_S + kc + bk + 8);
            acc0 = __builtin_amdgcn_wmma_f32_16x16x32_bf16(false, a.v, false, b0.v,
                                                           (short)0, acc0, false, false);
            acc1 = __builtin_amdgcn_wmma_f32_16x16x32_bf16(false, a.v, false, b1.v,
                                                           (short)0, acc1, false, false);
        }
    };

    // prologue
    stage_tile(0, 0);
    asm volatile("s_wait_asynccnt 0x0" ::: "memory");
    __syncthreads();

    for (int k0 = 0; k0 < Nn; k0 += 64) {
        const int cur = (k0 >> 6) & 1;
        if (k0 + 64 < Nn) stage_tile(k0 + 64, cur ^ 1);
        compute_tile(cur);
        asm volatile("s_wait_asynccnt 0x0" ::: "memory");
        __syncthreads();
    }

    // ---- deg reduction ----
    atomicAdd(&degs[sc + 0], degp[0]);
    atomicAdd(&degs[sc + 1], degp[1]);
    atomicAdd(&degs[sc + 2], degp[2]);
    atomicAdd(&degs[sc + 3], degp[3]);
    __syncthreads();

    // ---- scale msum -> mean (bf16) in LDS ----
    // C layout: VGPR g, lanes 0-15: M=g, N=lane; lanes 16-31: M=g+8, N=lane-16
    {
        const int mhalf = (lane >= 16) ? 8 : 0;
        #pragma unroll
        for (int g = 0; g < 8; ++g) {
            const int jl = Msub * 16 + g + mhalf;
            const float inv = 1.0f / fmaxf(degs[jl], 1.0f);
            meanT[jl * ME_S + f0a + ln15] = (__bf16)(acc0[g] * inv);
            meanT[jl * ME_S + f0b + ln15] = (__bf16)(acc1[g] * inv);
        }
    }
    __syncthreads();

    // ---- stage 2: out = relu(mean@Wmsg + x@Wupd + bias) ----
    const int u0a = f0a, u0b = f0b;
    const float bva = bias[u0a + ln15];
    const float bvb = bias[u0b + ln15];
    v8f o0, o1;
    #pragma unroll
    for (int g = 0; g < 8; ++g) { o0[g] = bva; o1[g] = bvb; }

    const size_t xh_row = ((size_t)batch * Nn + j0 + mrow) * Fn;
    #pragma unroll
    for (int fk = 0; fk < Fn; fk += 32) {
        V16 am, ax, bm0, bm1, bu0, bu1;
        am.q[0] = *(const uint4*)(meanT + mrow * ME_S + fk + klo);
        am.q[1] = *(const uint4*)(meanT + mrow * ME_S + fk + klo + 8);
        ax.q[0] = *(const uint4*)(xh + xh_row + fk + klo);
        ax.q[1] = *(const uint4*)(xh + xh_row + fk + klo + 8);

        const size_t wra = (size_t)(u0a + ln15) * Fn + fk + bk;
        const size_t wrb = (size_t)(u0b + ln15) * Fn + fk + bk;
        bm0.q[0] = *(const uint4*)(wTm + wra);
        bm0.q[1] = *(const uint4*)(wTm + wra + 8);
        bm1.q[0] = *(const uint4*)(wTm + wrb);
        bm1.q[1] = *(const uint4*)(wTm + wrb + 8);
        bu0.q[0] = *(const uint4*)(wTu + wra);
        bu0.q[1] = *(const uint4*)(wTu + wra + 8);
        bu1.q[0] = *(const uint4*)(wTu + wrb);
        bu1.q[1] = *(const uint4*)(wTu + wrb + 8);

        o0 = __builtin_amdgcn_wmma_f32_16x16x32_bf16(false, am.v, false, bm0.v,
                                                     (short)0, o0, false, false);
        o0 = __builtin_amdgcn_wmma_f32_16x16x32_bf16(false, ax.v, false, bu0.v,
                                                     (short)0, o0, false, false);
        o1 = __builtin_amdgcn_wmma_f32_16x16x32_bf16(false, am.v, false, bm1.v,
                                                     (short)0, o1, false, false);
        o1 = __builtin_amdgcn_wmma_f32_16x16x32_bf16(false, ax.v, false, bu1.v,
                                                     (short)0, o1, false, false);
    }

    // ---- relu + store ----
    {
        const int mhalf = (lane >= 16) ? 8 : 0;
        #pragma unroll
        for (int g = 0; g < 8; ++g) {
            const int row = j0 + Msub * 16 + g + mhalf;
            const size_t ob = ((size_t)batch * Nn + row) * Un;
            out[ob + u0a + ln15] = fmaxf(o0[g], 0.0f);
            out[ob + u0b + ln15] = fmaxf(o1[g], 0.0f);
        }
    }
}

// ---------------------------------------------------------------------------
extern "C" void kernel_launch(void* const* d_in, const int* in_sizes, int n_in,
                              void* d_out, int out_size, void* d_ws, size_t ws_size,
                              hipStream_t stream) {
    (void)in_sizes; (void)n_in; (void)out_size; (void)ws_size;
    const float* x    = (const float*)d_in[0];
    const int*   adj  = (const int*)d_in[1];
    const float* Wm   = (const float*)d_in[2];
    const float* Wu   = (const float*)d_in[3];
    const float* bias = (const float*)d_in[4];
    float* out = (float*)d_out;

    char* ws = (char*)d_ws;
    const size_t xh_bytes = (size_t)Bn * Nn * Fn * 2;   // 8 MB
    __bf16* xh  = (__bf16*)ws;
    __bf16* xT  = (__bf16*)(ws + xh_bytes);
    __bf16* wTm = (__bf16*)(ws + 2 * xh_bytes);
    __bf16* wTu = wTm + (size_t)Fn * Un;

    prep_x_kernel<<<Bn * (Nn / 32) * (Fn / 32), 256, 0, stream>>>(x, xh, xT);
    prep_w_kernel<<<(2 * Fn * Un) / 256, 256, 0, stream>>>(Wm, Wu, wTm, wTu);
    mpnn_main<<<Bn * (Nn / 32), 256, 0, stream>>>(adj, xh, xT, wTm, wTu, bias, out);
}
